// Deep_Root_Net_39264591020386
// MI455X (gfx1250) — matrix operand order repaired and checked by reading, not compiled, not run on hardware
//
#include <hip/hip_runtime.h>
#include <hip/hip_bf16.h>
#include <math.h>
#include <stdint.h>

typedef __attribute__((ext_vector_type(16))) _Float16     v16h;
typedef __attribute__((ext_vector_type(8)))  float        v8f;
typedef __attribute__((ext_vector_type(2)))  float        v2f;
typedef __attribute__((ext_vector_type(4)))  unsigned int v4u;
typedef __attribute__((ext_vector_type(8)))  int          v8i;
typedef __attribute__((ext_vector_type(4)))  int          v4i;

#if defined(__has_builtin)
#if __has_builtin(__builtin_amdgcn_tensor_load_to_lds) && __has_builtin(__builtin_amdgcn_s_wait_tensorcnt)
#define HAVE_TDM 1
#endif
#endif
#ifndef HAVE_TDM
#define HAVE_TDM 0
#endif

// ---------------------------------------------------------------------------
// Problem constants (match reference)
// ---------------------------------------------------------------------------
constexpr int   kTau   = 8;
constexpr int   kN     = 16;          // sensors
constexpr int   kBatch = 1024;
constexpr int   kM     = 3;           // sources
constexpr float kSlope = 0.5f;        // LeakyReLU
constexpr float kEps   = 1.0f;        // diagonal loading
constexpr int   kDeg   = 2 * kN - 2;  // 30 polynomial degree
constexpr int   kNCoef = 2 * kN - 1;  // 31 coefficients
constexpr float kInvPi = 0.31830988618379067f;

// d_out layout (floats; complex interleaved re,im), reference return order:
// DOA [B,3] | DOA_all [B,30] | roots_s[-1] [30]c | Rz [B,16,16]c | sorted [B,30]
constexpr size_t OFF_DOA   = 0;
constexpr size_t OFF_ALL   = OFF_DOA   + (size_t)kBatch * kM;
constexpr size_t OFF_ROOTS = OFF_ALL   + (size_t)kBatch * kDeg;
constexpr size_t OFF_RZ    = OFF_ROOTS + (size_t)2 * kDeg;
constexpr size_t OFF_SORT  = OFF_RZ    + (size_t)kBatch * kN * kN * 2;

// ---------------------------------------------------------------------------
// Complex helpers
// ---------------------------------------------------------------------------
struct cpx { float re, im; };
__device__ __forceinline__ cpx cadd(cpx a, cpx b) { return {a.re + b.re, a.im + b.im}; }
__device__ __forceinline__ cpx csub(cpx a, cpx b) { return {a.re - b.re, a.im - b.im}; }
__device__ __forceinline__ cpx cmul(cpx a, cpx b) {
    return {a.re * b.re - a.im * b.im, a.re * b.im + a.im * b.re};
}
__device__ __forceinline__ cpx cdiv(cpx a, cpx b) {
    float d = b.re * b.re + b.im * b.im + 1e-30f;
    return {(a.re * b.re + a.im * b.im) / d, (a.im * b.re - a.re * b.im) / d};
}

// ---------------------------------------------------------------------------
// Tensor Data Mover: 1-D tile (nelem f32, contiguous) from global -> LDS.
// D# per CDNA5 ISA: group0 = {count=1 | lds_addr |
// global_addr | type=2}; group1 = {data_size=4B, tensor_dim0=tile_dim0=nelem,
// tensor_dim1=1, tensor_dim0_stride=nelem}.  Issued once per wave (EXEC is
// ignored for TENSOR ops); completion via TENSORcnt.
// ---------------------------------------------------------------------------
__device__ __forceinline__ void tdm_load_f32_1d(uint32_t lds_byte_off,
                                                const float* gsrc, int nelem) {
#if HAVE_TDM
    const unsigned long long ga = (unsigned long long)(uintptr_t)gsrc;
    v4u g0;
    g0[0] = 1u;                                   // count=1, user descriptor
    g0[1] = lds_byte_off;                         // lds_addr
    g0[2] = (unsigned)(ga & 0xffffffffull);       // global_addr[31:0]
    g0[3] = (unsigned)((ga >> 32) & 0x01ffffffull) | (2u << 30);  // addr[56:32] | type=2
    const unsigned n = (unsigned)nelem;
    v8i g1;
    g1[0] = (int)(2u << 16);                      // workgroup_mask=0, data_size=4B
    g1[1] = (int)((n & 0xffffu) << 16);           // tensor_dim0[15:0]
    g1[2] = (int)(((n >> 16) & 0xffffu) | (1u << 16));  // dim0[31:16] | tensor_dim1=1
    g1[3] = (int)((n & 0xffffu) << 16);           // tile_dim0 = nelem
    g1[4] = 0;                                    // tile_dim1/2 unused
    g1[5] = (int)n;                               // tensor_dim0_stride[31:0]
    g1[6] = 0;
    g1[7] = 0;
    v4i z = {0, 0, 0, 0};
#if __clang_major__ >= 23
    v8i z8 = {0, 0, 0, 0, 0, 0, 0, 0};
    __builtin_amdgcn_tensor_load_to_lds(g0, g1, z, z, z8, 0);
#else
    __builtin_amdgcn_tensor_load_to_lds(g0, g1, z, z, 0);
#endif
    __builtin_amdgcn_s_wait_tensorcnt((short)0);  // s_wait_tensorcnt 0
#else
    (void)lds_byte_off; (void)gsrc; (void)nelem;
#endif
}

// ---------------------------------------------------------------------------
// Implicit-GEMM 2x2 conv / stride-1 transposed-conv via V_WMMA_F32_16X16X32_F16.
//   GEMM: M = OH*OW (spatial), N = C_out, K = 4*C_in (always a multiple of 32).
//   One wave computes a 16x16 output tile; 4 waves per block.
//   Intermediate activations are stored f16 (TIN/TOUT): WMMA consumes f16
//   anyway and the layers are bandwidth-bound, so this halves HBM/L2 traffic.
// A-fragment layout (ISA 7.12.2, 16-bit A 16x32): lane row = lane&15,
//   VGPR v<4 -> K = 2v + 8*(lane>>4), v>=4 -> K = 16 + 2(v-4) + 8*(lane>>4).
// B-fragment (32x16): lane col = lane&15, K same as A (table order).
// C/D (v8f): col = lane&15, row = r + 8*(lane>>4).
// ---------------------------------------------------------------------------
template <typename TIN, typename TOUT, int CIN, int COUT, int IH, int IW,
          int OH, int OW, bool TRANSPOSED, bool RELU>
__global__ __launch_bounds__(128) void conv2x2_wmma(const TIN* __restrict__ x,
                                                    const float* __restrict__ w,
                                                    const float* __restrict__ bias,
                                                    TOUT* __restrict__ y) {
    constexpr int K    = CIN * 4;
    constexpr int MTOT = OH * OW;
    constexpr int TM   = (MTOT + 15) / 16;
    constexpr int TN   = (COUT + 15) / 16;
    constexpr int PAD  = TRANSPOSED ? 1 : 0;

    const int wave = threadIdx.x >> 5;
    const int lane = threadIdx.x & 31;
    const int tile = blockIdx.x * 4 + wave;
    if (tile >= kBatch * TM * TN) return;   // whole wave exits: EXEC stays full

    const int b    = tile / (TM * TN);
    const int tm   = (tile % (TM * TN)) / TN;
    const int tn   = tile % TN;
    const int half = lane >> 4;
    const int l15  = lane & 15;
    const int mrow = tm * 16 + l15;         // A row (spatial index)
    const int ncol = tn * 16 + l15;         // B col (output channel)

    const TIN* xb = x + (size_t)b * CIN * IH * IW;
    __builtin_prefetch(xb, 0, 1);           // global_prefetch_b8

    v8f acc = {};
    for (int k0 = 0; k0 < K; k0 += 32) {
        v16h a;
        #pragma unroll
        for (int j = 0; j < 16; ++j) {
            const int v   = j >> 1, pos = j & 1;
            const int kk  = k0 + ((v < 4) ? 2 * v : 16 + 2 * (v - 4)) + 8 * half + pos;
            float val = 0.f;
            if (mrow < MTOT) {
                const int c  = kk >> 2;
                const int kh = (kk >> 1) & 1;
                const int kw = kk & 1;
                const int oh = mrow / OW, ow = mrow % OW;
                const int ih = oh + kh - PAD, iw = ow + kw - PAD;
                if (ih >= 0 && ih < IH && iw >= 0 && iw < IW)
                    val = (float)xb[((size_t)c * IH + ih) * IW + iw];
            }
            a[j] = (_Float16)val;
        }
        v16h bf;
        #pragma unroll
        for (int j = 0; j < 16; ++j) {
            const int v  = j >> 1, pos = j & 1;
            const int kk = k0 + ((v < 4) ? 2 * v : 16 + 2 * (v - 4)) + 8 * half + pos;
            float val = 0.f;
            if (ncol < COUT) {
                const int c  = kk >> 2;
                const int kh = (kk >> 1) & 1;
                const int kw = kk & 1;
                if (!TRANSPOSED)  // w[COUT][CIN][2][2]
                    val = w[(((size_t)ncol * CIN + c) * 2 + kh) * 2 + kw];
                else              // w[CIN][COUT][2][2], flipped spatially
                    val = w[(((size_t)c * COUT + ncol) * 2 + (1 - kh)) * 2 + (1 - kw)];
            }
            bf[j] = (_Float16)val;
        }
        acc = __builtin_amdgcn_wmma_f32_16x16x32_f16(false, a, false, bf,
                                                     (short)0, acc, false, false);
    }

    if (ncol < COUT) {
        const float bv = bias[ncol];
        TOUT* yb = y + (size_t)b * COUT * MTOT;
        #pragma unroll
        for (int r = 0; r < 8; ++r) {
            const int m = tm * 16 + r + 8 * half;
            if (m < MTOT) {
                float v = acc[r] + bv;
                if (RELU) v = (v >= 0.f) ? v : kSlope * v;
                yb[(size_t)ncol * MTOT + m] = (TOUT)v;
            }
        }
    }
}

// ---------------------------------------------------------------------------
// Gramian: Rz = K^H K + eps*I (complex 16x16 per batch), full f32 precision via
// V_WMMA_F32_16X16X4_F32.  Re = Kr^T Kr + Ki^T Ki;  Im = Kr^T Ki - Ki^T Kr.
// Per-lane the A (transposed) and B fragments are the same LDS loads.
// 4 waves / block, one batch per wave; per-wave rx tile staged via TDM.
// ---------------------------------------------------------------------------
__global__ __launch_bounds__(128) void gramian_wmma(const float* __restrict__ rx,
                                                    float* __restrict__ rz,
                                                    float* __restrict__ out) {
    __shared__ float Kbuf[4][512];   // [wave][ Kr(256) | Ki(256) ] = rx slice layout
    const int wave = threadIdx.x >> 5;
    const int lane = threadIdx.x & 31;
    const int b    = blockIdx.x * 4 + wave;

    const float* p = rx + (size_t)b * 2 * kN * kN;
#if HAVE_TDM
    tdm_load_f32_1d((uint32_t)(uintptr_t)&Kbuf[wave][0], p, 512);
#else
    for (int t = lane; t < 512; t += 32) Kbuf[wave][t] = p[t];
#endif
    __syncthreads();
    const float* Kr = &Kbuf[wave][0];
    const float* Ki = &Kbuf[wave][256];

    const int half = lane >> 4, l15 = lane & 15;
    v8f cre = {}, p1 = {}, p2 = {};
    #pragma unroll
    for (int i0 = 0; i0 < 16; i0 += 4) {
        v2f ar, ai;   // f32 A 16x4 layout: lane row = l15, K = 2*half + v
        ar[0] = Kr[(i0 + 2 * half + 0) * 16 + l15];
        ar[1] = Kr[(i0 + 2 * half + 1) * 16 + l15];
        ai[0] = Ki[(i0 + 2 * half + 0) * 16 + l15];
        ai[1] = Ki[(i0 + 2 * half + 1) * 16 + l15];
        cre = __builtin_amdgcn_wmma_f32_16x16x4_f32(false, ar, false, ar, (short)0, cre, false, false);
        cre = __builtin_amdgcn_wmma_f32_16x16x4_f32(false, ai, false, ai, (short)0, cre, false, false);
        p1  = __builtin_amdgcn_wmma_f32_16x16x4_f32(false, ar, false, ai, (short)0, p1,  false, false);
        p2  = __builtin_amdgcn_wmma_f32_16x16x4_f32(false, ai, false, ar, (short)0, p2,  false, false);
    }
    #pragma unroll
    for (int r = 0; r < 8; ++r) {
        const int m = r + 8 * half, n = l15;
        const float re = cre[r] + ((m == n) ? kEps : 0.f);
        const float im = p1[r] - p2[r];
        const size_t idx = (size_t)b * 256 + m * 16 + n;
        rz[2 * idx]     = re;
        rz[2 * idx + 1] = im;
        out[OFF_RZ + 2 * idx]     = re;
        out[OFF_RZ + 2 * idx + 1] = im;
    }
}

// ---------------------------------------------------------------------------
// Per-batch root-MUSIC tail: complex Hermitian Jacobi eig of Rz, noise
// projector F = Un Un^H via f32 WMMA, diagonal sums -> polynomial, Durand-
// Kerner roots (replaces the 30x30 companion eig), sort, DOA.
// One wave (32 threads) per batch; Rz staged via TDM; all state in LDS.
// ---------------------------------------------------------------------------
__global__ __launch_bounds__(32) void eig_music(const float* __restrict__ rz,
                                                float* __restrict__ out) {
    __shared__ cpx   A[16][16];    // contiguous 512 floats, matches rz slice
    __shared__ cpx   V[16][16];
    __shared__ float Ur[16][16], Ui[16][16];
    __shared__ cpx   Fm[16][16];
    __shared__ cpx   coef[kNCoef];
    __shared__ cpx   zbuf[kDeg];
    __shared__ float dists[kDeg];
    __shared__ int   order[16];
    __shared__ int   o2[kDeg];
    __shared__ float rot[4];  // c, s.re, s.im, skip-flag

    const int b    = blockIdx.x;
    const int lane = threadIdx.x;

    const float* src = rz + (size_t)b * 2 * kN * kN;
#if HAVE_TDM
    tdm_load_f32_1d((uint32_t)(uintptr_t)&A[0][0], src, 512);
    for (int t = lane; t < 256; t += 32) {
        const int i = t >> 4, j = t & 15;
        V[i][j] = {(i == j) ? 1.f : 0.f, 0.f};
    }
#else
    for (int t = lane; t < 256; t += 32) {
        const int i = t >> 4, j = t & 15;
        A[i][j] = {src[2 * t], src[2 * t + 1]};
        V[i][j] = {(i == j) ? 1.f : 0.f, 0.f};
    }
#endif
    __syncthreads();

    // --- cyclic complex Jacobi (Rz Hermitian PSD) -------------------------
    for (int sweep = 0; sweep < 6; ++sweep) {
        for (int p = 0; p < 15; ++p) {
            for (int q = p + 1; q < 16; ++q) {
                if (lane == 0) {
                    const cpx apq = A[p][q];
                    const float m = sqrtf(apq.re * apq.re + apq.im * apq.im);
                    if (m < 1e-12f) {
                        rot[3] = 1.f;
                    } else {
                        const float app = A[p][p].re, aqq = A[q][q].re;
                        const float tau = (aqq - app) / (2.f * m);
                        const float t = (tau >= 0.f ? 1.f : -1.f) /
                                        (fabsf(tau) + sqrtf(1.f + tau * tau));
                        const float c = rsqrtf(1.f + t * t);
                        const float sr = t * c;
                        rot[0] = c;
                        rot[1] = sr * apq.re / m;
                        rot[2] = sr * apq.im / m;
                        rot[3] = 0.f;
                    }
                }
                __syncthreads();
                if (rot[3] == 0.f) {   // uniform across wave
                    const float c = rot[0];
                    const cpx s = {rot[1], rot[2]};
                    if (lane < 16) {   // column update: M = A*G, V = V*G
                        const int i = lane;
                        const cpx aip = A[i][p], aiq = A[i][q];
                        A[i][p] = {c * aip.re + s.re * aiq.re - s.im * aiq.im,
                                   c * aip.im + s.re * aiq.im + s.im * aiq.re};
                        A[i][q] = {c * aiq.re - (s.re * aip.re + s.im * aip.im),
                                   c * aiq.im - (s.re * aip.im - s.im * aip.re)};
                        const cpx vip = V[i][p], viq = V[i][q];
                        V[i][p] = {c * vip.re + s.re * viq.re - s.im * viq.im,
                                   c * vip.im + s.re * viq.im + s.im * viq.re};
                        V[i][q] = {c * viq.re - (s.re * vip.re + s.im * vip.im),
                                   c * viq.im - (s.re * vip.im - s.im * vip.re)};
                    }
                    __syncthreads();
                    if (lane < 16) {   // row update: A = G^H * M
                        const int j = lane;
                        const cpx apj = A[p][j], aqj = A[q][j];
                        A[p][j] = {c * apj.re + s.re * aqj.re + s.im * aqj.im,
                                   c * apj.im + s.re * aqj.im - s.im * aqj.re};
                        A[q][j] = {c * aqj.re - (s.re * apj.re - s.im * apj.im),
                                   c * aqj.im - (s.re * apj.im + s.im * apj.re)};
                    }
                }
                __syncthreads();
            }
        }
    }

    // --- sort eigenvalues desc, build noise-subspace matrix ---------------
    if (lane < 16) order[lane] = lane;
    __syncthreads();
    if (lane == 0) {
        for (int i = 1; i < 16; ++i) {
            const int key = order[i];
            const float kv = fabsf(A[key][key].re);
            int j = i - 1;
            while (j >= 0 && fabsf(A[order[j]][order[j]].re) < kv) {
                order[j + 1] = order[j];
                --j;
            }
            order[j + 1] = key;
        }
    }
    __syncthreads();
    if (lane < 16) {   // signal columns (top kM) zeroed -> F = Un Un^H exactly
        const int i = lane;
        for (int k = 0; k < 16; ++k) {
            if (k >= kM) {
                const int sc = order[k];
                Ur[i][k] = V[i][sc].re;
                Ui[i][k] = V[i][sc].im;
            } else {
                Ur[i][k] = 0.f;
                Ui[i][k] = 0.f;
            }
        }
    }
    __syncthreads();

    // --- F = U U^H via f32 WMMA: Re = Ur Ur^T + Ui Ui^T, Im = Ui Ur^T - Ur Ui^T
    {
        const int half = lane >> 4, l15 = lane & 15;
        v8f fre = {}, t1 = {}, t2 = {};
        #pragma unroll
        for (int k0 = 0; k0 < 16; k0 += 4) {
            v2f ur, ui;
            ur[0] = Ur[l15][k0 + 2 * half + 0];
            ur[1] = Ur[l15][k0 + 2 * half + 1];
            ui[0] = Ui[l15][k0 + 2 * half + 0];
            ui[1] = Ui[l15][k0 + 2 * half + 1];
            fre = __builtin_amdgcn_wmma_f32_16x16x4_f32(false, ur, false, ur, (short)0, fre, false, false);
            fre = __builtin_amdgcn_wmma_f32_16x16x4_f32(false, ui, false, ui, (short)0, fre, false, false);
            t1  = __builtin_amdgcn_wmma_f32_16x16x4_f32(false, ui, false, ur, (short)0, t1,  false, false);
            t2  = __builtin_amdgcn_wmma_f32_16x16x4_f32(false, ur, false, ui, (short)0, t2,  false, false);
        }
        #pragma unroll
        for (int r = 0; r < 8; ++r) {
            const int m = r + 8 * half;
            Fm[m][l15] = {fre[r], t1[r] - t2[r]};
        }
    }
    __syncthreads();

    // --- sum of diagonals -> polynomial coefficients ----------------------
    if (lane < kNCoef) {
        const int off = lane - (kN - 1);
        cpx s = {0.f, 0.f};
        for (int i = 0; i < 16; ++i) {
            const int j = i + off;
            if (j >= 0 && j < 16) s = cadd(s, Fm[i][j]);
        }
        coef[lane] = s;
    }
    __syncthreads();
    {   // normalize to monic (matches companion construction -coeff[1:]/coeff[0])
        const cpx c0 = coef[0];
        const cpx cd = (lane < kNCoef) ? coef[lane] : cpx{0.f, 0.f};
        __syncthreads();
        if (lane < kNCoef) coef[lane] = cdiv(cd, c0);
        __syncthreads();
    }

    // --- Durand-Kerner roots of z^30 + c1 z^29 + ... + c30 ----------------
    if (lane < kDeg) {   // init z_i = (0.4 + 0.9i)^(i+1)
        const float m0 = 0.98488578f, a0 = 1.1525719972156676f;
        const float rr = powf(m0, (float)(lane + 1));
        zbuf[lane] = {rr * cosf(a0 * (lane + 1)), rr * sinf(a0 * (lane + 1))};
    }
    __syncthreads();
    for (int it = 0; it < 80; ++it) {
        cpx zn = {0.f, 0.f};
        if (lane < kDeg) {
            const cpx zi = zbuf[lane];
            cpx pv = {1.f, 0.f};
            for (int k = 1; k < kNCoef; ++k) pv = cadd(cmul(pv, zi), coef[k]);
            cpx den = {1.f, 0.f};
            for (int j = 0; j < kDeg; ++j)
                if (j != lane) den = cmul(den, csub(zi, zbuf[j]));
            zn = csub(zi, cdiv(pv, den));
        }
        __syncthreads();
        if (lane < kDeg) zbuf[lane] = zn;
        __syncthreads();
    }

    // --- outputs ----------------------------------------------------------
    if (lane < kDeg) {
        const cpx z = zbuf[lane];
        const float mag = sqrtf(z.re * z.re + z.im * z.im);
        dists[lane] = fabsf(mag - 1.f);
        out[OFF_ALL + (size_t)b * kDeg + lane] = asinf(atan2f(z.im, z.re) * kInvPi);
    }
    __syncthreads();
    if (lane == 0) {
        for (int k = 0; k < kDeg; ++k) o2[k] = k;
        for (int i = 1; i < kDeg; ++i) {    // stable insertion sort by |(|z|-1)|
            const int key = o2[i];
            const float kv = dists[key];
            int j = i - 1;
            while (j >= 0 && dists[o2[j]] > kv) { o2[j + 1] = o2[j]; --j; }
            o2[j + 1] = key;
        }
        int cnt = 0;   // first M inside unit circle (stable), then outside
        for (int pass = 0; pass < 2 && cnt < kM; ++pass) {
            for (int k = 0; k < kDeg && cnt < kM; ++k) {
                const cpx z = zbuf[o2[k]];
                const float mag = sqrtf(z.re * z.re + z.im * z.im);
                const bool inside = (mag - 1.f) < 0.f;
                if ((pass == 0) == inside)
                    out[OFF_DOA + (size_t)b * kM + cnt++] =
                        asinf(atan2f(z.im, z.re) * kInvPi);
            }
        }
    }
    __syncthreads();
    if (lane < kDeg) {
        const cpx z = zbuf[o2[lane]];
        out[OFF_SORT + (size_t)b * kDeg + lane] = asinf(atan2f(z.im, z.re) * kInvPi);
        if (b == kBatch - 1) {   // roots_s of last batch element
            out[OFF_ROOTS + 2 * lane]     = z.re;
            out[OFF_ROOTS + 2 * lane + 1] = z.im;
        }
    }
}

// ---------------------------------------------------------------------------
// Host entry.  Workspace (bytes): f16 ping (act1/act3/act5) | f16 pong
// (act2/act4) | f32 rx [B,32,16] | f32 rz [B,16,16]c  ->  ~81 MB total.
// ---------------------------------------------------------------------------
extern "C" void kernel_launch(void* const* d_in, const int* in_sizes, int n_in,
                              void* d_out, int out_size, void* d_ws, size_t ws_size,
                              hipStream_t stream) {
    (void)in_sizes; (void)n_in; (void)out_size; (void)ws_size;
    const float* x   = (const float*)d_in[0];
    // d_in[1] is M (==3), compile-time constant here
    const float* c1w = (const float*)d_in[2];
    const float* c1b = (const float*)d_in[3];
    const float* c2w = (const float*)d_in[4];
    const float* c2b = (const float*)d_in[5];
    const float* c3w = (const float*)d_in[6];
    const float* c3b = (const float*)d_in[7];
    const float* d2w = (const float*)d_in[8];
    const float* d2b = (const float*)d_in[9];
    const float* d3w = (const float*)d_in[10];
    const float* d3b = (const float*)d_in[11];
    const float* d4w = (const float*)d_in[12];
    const float* d4b = (const float*)d_in[13];
    float* out = (float*)d_out;

    char* wsb = (char*)d_ws;
    _Float16* actA = (_Float16*)wsb;                         // act1/act3/act5
    _Float16* actB = (_Float16*)(wsb + (size_t)kBatch * 64 * 29 * 13 * sizeof(_Float16));
    float*    rx   = (float*)(wsb + (size_t)kBatch * 64 * 29 * 13 * sizeof(_Float16)
                                  + (size_t)kBatch * 32 * 30 * 14 * sizeof(_Float16));
    float*    rz   = rx + (size_t)kBatch * 2 * kN * kN;

    const dim3 blk(128);
    auto blocks = [](int mtot, int cout) {
        const int tiles = kBatch * ((mtot + 15) / 16) * ((cout + 15) / 16);
        return dim3((tiles + 3) / 4);
    };
    conv2x2_wmma<float,    _Float16,  8, 16, 32, 16, 31, 15, false, true >
        <<<blocks(31 * 15, 16), blk, 0, stream>>>(x,    c1w, c1b, actA);
    conv2x2_wmma<_Float16, _Float16, 16, 32, 31, 15, 30, 14, false, true >
        <<<blocks(30 * 14, 32), blk, 0, stream>>>(actA, c2w, c2b, actB);
    conv2x2_wmma<_Float16, _Float16, 32, 64, 30, 14, 29, 13, false, true >
        <<<blocks(29 * 13, 64), blk, 0, stream>>>(actB, c3w, c3b, actA);
    conv2x2_wmma<_Float16, _Float16, 64, 32, 29, 13, 30, 14, true,  true >
        <<<blocks(30 * 14, 32), blk, 0, stream>>>(actA, d2w, d2b, actB);
    conv2x2_wmma<_Float16, _Float16, 32, 16, 30, 14, 31, 15, true,  true >
        <<<blocks(31 * 15, 16), blk, 0, stream>>>(actB, d3w, d3b, actA);
    conv2x2_wmma<_Float16, float,    16,  1, 31, 15, 32, 16, true,  false>
        <<<blocks(32 * 16,  1), blk, 0, stream>>>(actA, d4w, d4b, rx);

    gramian_wmma<<<dim3(kBatch / 4), blk, 0, stream>>>(rx, rz, out);
    eig_music<<<dim3(kBatch), dim3(32), 0, stream>>>(rz, out);
}
